// HyperspectralCmKANLayer_34428457845128
// MI455X (gfx1250) — matrix operand order repaired and checked by reading, not compiled, not run on hardware
//
#include <hip/hip_runtime.h>
#include <hip/hip_bf16.h>

// ---------------- problem constants ----------------
#define IN_CH   31
#define OUT_CH  31
#define NB      8          // G + K = 5 + 3
#define NPIX    16384      // 128*128, B=1
#define HH      128
#define WW      128
#define KREAL   279        // IN_CH * 9
#define KPAD    288        // padded to multiple of 32 (9 WMMA K-steps)
#define KSTEPS  9
#define NPARAM  9610
#define NPAD    9664       // 151 * 64 (pad param dim to N-tile of 64)
#define NT64    151
#define COEF_LEN 7688      // 31*31*8
#define UNIV_OFF 7688
#define RES_OFF  8649      // 7688 + 961

typedef __bf16 v16bf __attribute__((ext_vector_type(16)));
typedef float  v8f   __attribute__((ext_vector_type(8)));

union ABfrag { uint4 u4[2]; v16bf v; };
union CDfrag { v8f v; float4 f4[2]; float f[8]; };

// RNE float -> bf16 bits
__device__ __forceinline__ unsigned short f2bf(float f) {
    unsigned int u = __float_as_uint(f);
    unsigned int lsb = (u >> 16) & 1u;
    u += 0x7fffu + lsb;
    return (unsigned short)(u >> 16);
}

__device__ __forceinline__ float knot(int j) {
    return -1.0f + (float)(j - 3) * 0.4f;   // h = 2/5
}

// A fragment (16x32 bf16): lanes 0-15 hold K {0..15}, lanes 16-31 hold {16..31}
__device__ __forceinline__ void load_a(ABfrag& f, const unsigned short* arow,
                                       int kk, int halfsel) {
    const uint4* ap = (const uint4*)(arow + kk * 32 + halfsel * 16);
    f.u4[0] = ap[0];   // +0  elements
    f.u4[1] = ap[1];   // +8 elements (= +16 bytes)
}

// B fragment (32x16 bf16): lane holds N=rlane, contiguous 16 K at halfsel*16
__device__ __forceinline__ void load_b(ABfrag& f, const unsigned short* brow,
                                       int kk, int halfsel) {
    const uint4* bp = (const uint4*)(brow + kk * 32 + halfsel * 16);
    f.u4[0] = bp[0];
    f.u4[1] = bp[1];
}

// ---------------- prep: gen_w (9610x279 f32) -> bf16 [NPAD][KPAD], zero padded ----------------
__global__ void k_prep_w(const float* __restrict__ gen_w, unsigned short* __restrict__ w16) {
    int t = blockIdx.x * blockDim.x + threadIdx.x;
    if (t >= NPAD * KPAD) return;
    int n = t / KPAD;
    int k = t - n * KPAD;
    float v = 0.0f;
    if (n < NPARAM && k < KREAL) v = gen_w[(size_t)n * KREAL + k];
    w16[t] = f2bf(v);
}

// ---------------- prep: im2col of x -> bf16 [NPIX][KPAD] (SAME 3x3, cross-correlation) -------
__global__ void k_prep_im(const float* __restrict__ x, unsigned short* __restrict__ im) {
    int t = blockIdx.x * blockDim.x + threadIdx.x;
    if (t >= NPIX * KPAD) return;
    int pix = t / KPAD;
    int k   = t - pix * KPAD;
    float v = 0.0f;
    if (k < KREAL) {
        int i  = k / 9;
        int r  = k - i * 9;
        int ky = r / 3;
        int kx = r - ky * 3;
        int h = pix / WW;
        int w = pix - h * WW;
        int hh = h + ky - 1;
        int ww = w + kx - 1;
        if (hh >= 0 && hh < HH && ww >= 0 && ww < WW)
            v = x[(size_t)i * NPIX + hh * WW + ww];
    }
    im[t] = f2bf(v);
}

// ---------------- GEMM: w[n][m] (transposed) = im2col(m,:) x gen_w(n,:)^T via bf16 WMMA ------
// One wave computes a 16(M) x 64(N) tile. Register double-buffered software pipeline,
// enforced with sched_barrier(0) fences so the scheduler cannot sink the kk+1 prefetch
// loads past the kk WMMA block (which would collapse the buffers and force loadcnt==0
// drains before every WMMA, as observed in rounds 0-1).
__global__ void __launch_bounds__(128)
k_gemm(const unsigned short* __restrict__ A,   // im2col [NPIX][KPAD]
       const unsigned short* __restrict__ Bw,  // weights [NPAD][KPAD]
       float* __restrict__ Wt,                 // out: [NPAD][chStride]
       int m_off, int Mtiles, int chStride) {
    int wave = blockIdx.x * (blockDim.x >> 5) + (threadIdx.x >> 5);
    int lane = threadIdx.x & 31;
    if (wave >= Mtiles * NT64) return;
    int mt = wave / NT64;
    int nt = wave - mt * NT64;

    int halfsel = lane >> 4;     // 0: lanes 0-15, 1: lanes 16-31
    int rlane   = lane & 15;

    int pix = m_off + mt * 16 + rlane;                     // A row (pixel)
    const unsigned short* arow  = A + (size_t)pix * KPAD;
    const unsigned short* brow0 = Bw + (size_t)(nt * 64 + rlane) * KPAD;

    CDfrag c[4];
#pragma unroll
    for (int s = 0; s < 4; ++s)
#pragma unroll
        for (int e = 0; e < 8; ++e) c[s].f[e] = 0.0f;

    ABfrag a0, a1;
    ABfrag b0[4], b1[4];

    // prologue: K-step 0 into buffer 0
    load_a(a0, arow, 0, halfsel);
#pragma unroll
    for (int s = 0; s < 4; ++s)
        load_b(b0[s], brow0 + (size_t)(s * 16) * KPAD, 0, halfsel);

#pragma unroll
    for (int kk = 0; kk < KSTEPS; ++kk) {
        ABfrag& ac = (kk & 1) ? a1 : a0;
        ABfrag* bc = (kk & 1) ? b1 : b0;
        ABfrag& an = (kk & 1) ? a0 : a1;
        ABfrag* bn = (kk & 1) ? b0 : b1;

        if (kk + 1 < KSTEPS) {   // prefetch next K-step into the alternate buffer
            load_a(an, arow, kk + 1, halfsel);
#pragma unroll
            for (int s = 0; s < 4; ++s)
                load_b(bn[s], brow0 + (size_t)(s * 16) * KPAD, kk + 1, halfsel);
        }

        // Fence: prefetch loads above must be issued before the WMMAs below,
        // keeping both fragment buffers live -> distinct VGPRs -> partial waits.
        __builtin_amdgcn_sched_barrier(0);

#pragma unroll
        for (int s = 0; s < 4; ++s)
            c[s].v = __builtin_amdgcn_wmma_f32_16x16x32_bf16(
                false, ac.v, false, bc[s].v, (short)0, c[s].v, false, false);

        __builtin_amdgcn_sched_barrier(0);
    }

    // D fragment: element r -> (M = 8*halfsel + r, N = rlane). Store transposed w[n][m].
    int mbase = mt * 16 + 8 * halfsel;
#pragma unroll
    for (int s = 0; s < 4; ++s) {
        int n = nt * 64 + s * 16 + rlane;
        float* dst = Wt + (size_t)n * chStride + mbase;
        ((float4*)dst)[0] = c[s].f4[0];
        ((float4*)dst)[1] = c[s].f4[1];
    }
}

// ---------------- consume: per-pixel KAN evaluation ----------------
__global__ void __launch_bounds__(256)
k_consume(const float* __restrict__ x,
          const float* __restrict__ Wt,      // [NPAD][chStride] conv results (no bias)
          const float* __restrict__ gen_b,
          float* __restrict__ out,
          int m_off, int chunkN, int chStride) {
    int t = blockIdx.x * blockDim.x + threadIdx.x;
    if (t >= chunkN) return;
    int pix = m_off + t;

    float acc[OUT_CH];
#pragma unroll
    for (int o = 0; o < OUT_CH; ++o) acc[o] = 0.0f;

    const float h = 0.4f;

    for (int i = 0; i < IN_CH; ++i) {
        float xv = x[(size_t)i * NPIX + pix];

        // Cox-de Boor: order-0 over 11 intervals, raise to order 3 -> 8 bases
        float b[11];
#pragma unroll
        for (int j = 0; j < 11; ++j)
            b[j] = (xv >= knot(j) && xv < knot(j + 1)) ? 1.0f : 0.0f;
#pragma unroll
        for (int p = 1; p <= 3; ++p) {
            float inv = 1.0f / ((float)p * h);
#pragma unroll 10
            for (int j = 0; j <= 10 - p; ++j) {
                b[j] = (xv - knot(j)) * inv * b[j] +
                       (knot(j + p + 1) - xv) * inv * b[j + 1];
            }
        }
        float silu = xv / (1.0f + __expf(-xv));

        int pu = UNIV_OFF + i * OUT_CH;
        int pr = RES_OFF + i * OUT_CH;
        for (int o = 0; o < OUT_CH; ++o) {
            int pc = (i * OUT_CH + o) * NB;
            float s = 0.0f;
#pragma unroll
            for (int d = 0; d < NB; ++d) {
                float cv = Wt[(size_t)(pc + d) * chStride + t] + gen_b[pc + d];
                s += cv * b[d];
            }
            float u = Wt[(size_t)(pu + o) * chStride + t] + gen_b[pu + o];
            float r = Wt[(size_t)(pr + o) * chStride + t] + gen_b[pr + o];
            acc[o] += u * s + silu * r;
        }
    }

#pragma unroll
    for (int o = 0; o < OUT_CH; ++o)
        out[(size_t)o * NPIX + pix] = acc[o];
}

// ---------------- launch ----------------
extern "C" void kernel_launch(void* const* d_in, const int* in_sizes, int n_in,
                              void* d_out, int out_size, void* d_ws, size_t ws_size,
                              hipStream_t stream) {
    const float* x     = (const float*)d_in[0];
    const float* gen_w = (const float*)d_in[1];
    const float* gen_b = (const float*)d_in[2];
    float* out = (float*)d_out;
    char* ws = (char*)d_ws;

    size_t off_w16 = 0;
    size_t sz_w16  = (size_t)NPAD * KPAD * sizeof(unsigned short);
    size_t off_im  = (off_w16 + sz_w16 + 255) & ~(size_t)255;
    size_t sz_im   = (size_t)NPIX * KPAD * sizeof(unsigned short);
    size_t off_buf = (off_im + sz_im + 255) & ~(size_t)255;

    size_t avail = (ws_size > off_buf) ? (ws_size - off_buf) : 0;
    long long chMax = (long long)(avail / ((size_t)NPAD * sizeof(float)));
    int CH = (int)((chMax / 16) * 16);
    if (CH < 16) CH = 16;             // minimal chunk (assumes ws covers >= 1 tile row)
    if (CH > 4096) CH = 4096;         // keep w-slab (<=158 MB) resident in the 192 MB L2

    unsigned short* w16 = (unsigned short*)(ws + off_w16);
    unsigned short* im  = (unsigned short*)(ws + off_im);
    float*          wbf = (float*)(ws + off_buf);

    {
        int total = NPAD * KPAD;
        k_prep_w<<<(total + 255) / 256, 256, 0, stream>>>(gen_w, w16);
    }
    {
        int total = NPIX * KPAD;
        k_prep_im<<<(total + 255) / 256, 256, 0, stream>>>(x, im);
    }

    for (int m_off = 0; m_off < NPIX; m_off += CH) {
        int chunkN = NPIX - m_off;
        if (chunkN > CH) chunkN = CH;
        int Mtiles = chunkN / 16;     // NPIX and CH are multiples of 16
        int waves  = Mtiles * NT64;
        int blocks = (waves + 3) / 4; // 4 waves / 128-thread block
        k_gemm<<<blocks, 128, 0, stream>>>(im, w16, wbf, m_off, Mtiles, CH);
        k_consume<<<(chunkN + 255) / 256, 256, 0, stream>>>(
            x, wbf, gen_b, out, m_off, chunkN, CH);
    }
}
